// ColumnGNN_60232621359199
// MI455X (gfx1250) — compile-verified, matrix operand order checked by reading, not run on hardware
//
#include <hip/hip_runtime.h>
#include <hip/hip_bf16.h>

typedef _Float16 half8  __attribute__((ext_vector_type(8)));
typedef _Float16 half16 __attribute__((ext_vector_type(16)));
typedef float    v8f    __attribute__((ext_vector_type(8)));

#define HH 128   // hidden
#define GG 64    // graphs
#define POOL_ROWS 16

// ---------------- elementwise / prep kernels ----------------

__global__ void k_zero_i32(int* p, int n) {
    int i = blockIdx.x * blockDim.x + threadIdx.x;
    if (i < n) p[i] = 0;
}

__global__ void k_f32_to_f16(const float* __restrict__ in, _Float16* __restrict__ out, int n) {
    int i = blockIdx.x * blockDim.x + threadIdx.x;
    if (i < n) out[i] = (_Float16)in[i];
}

// W [K][Nc] f32 row-major  ->  Wt [Nc][K] f16  (so B-fragment reads are contiguous in K)
__global__ void k_transpose_f16(const float* __restrict__ W, _Float16* __restrict__ Wt, int K, int Nc) {
    int i = blockIdx.x * blockDim.x + threadIdx.x;
    if (i < K * Nc) {
        int k = i / Nc, n = i % Nc;
        Wt[n * K + k] = (_Float16)W[i];
    }
}

__global__ void k_deg_accum(const int* __restrict__ dst, int* __restrict__ deg, int e) {
    int i = blockIdx.x * blockDim.x + threadIdx.x;
    if (i < e) atomicAdd(&deg[dst[i]], 1);
}

__global__ void k_make_dinv(const int* __restrict__ deg, float* __restrict__ dinv, int n) {
    int i = blockIdx.x * blockDim.x + threadIdx.x;
    if (i < n) dinv[i] = rsqrtf((float)(deg[i] + 1));   // +1 = self loop
}

// agg = hw * dinv^2  (self-loop contribution; also initializes agg, no pre-zero needed)
__global__ void k_self_loop(const float* __restrict__ hw, const float* __restrict__ dinv,
                            float* __restrict__ agg, int n) {
    int i = blockIdx.x * blockDim.x + threadIdx.x;
    if (i < n * HH) {
        float w = dinv[i >> 7];
        agg[i] = hw[i] * w * w;
    }
}

// ---------------- WMMA GEMM:  C[n][128] = A[n][K] (f16) x Wt[128][K]^T ----------------
// block = 256 threads = 8 waves; block owns 16 rows, wave w owns columns [16w,16w+16)
template<int K>
__global__ __launch_bounds__(256) void k_gemm_wmma(const _Float16* __restrict__ A,
                                                   const _Float16* __restrict__ Bt,
                                                   float* __restrict__ C, int nrows) {
    const int lane  = threadIdx.x & 31;
    const int wave  = threadIdx.x >> 5;
    const int m0    = blockIdx.x * 16;
    const int n0    = wave * 16;
    const int hi    = lane >> 4;          // lane half (0/1)
    const int lr    = lane & 15;
    const int kb    = hi * 8;             // K sub-offset per ISA 16-bit A layout
    int m = m0 + lr; if (m >= nrows) m = nrows - 1;

    const _Float16* arow = A  + (size_t)m * K;
    const _Float16* brow = Bt + (size_t)(n0 + lr) * K;

    v8f acc = {};
#pragma unroll
    for (int k0 = 0; k0 < K; k0 += 32) {
        half8 a_lo = *(const half8*)(arow + k0 + kb);
        half8 a_hi = *(const half8*)(arow + k0 + 16 + kb);
        half8 b_lo = *(const half8*)(brow + k0 + kb);
        half8 b_hi = *(const half8*)(brow + k0 + 16 + kb);
        half16 af, bf;
#pragma unroll
        for (int i = 0; i < 8; ++i) {
            af[i] = a_lo[i]; af[8 + i] = a_hi[i];
            bf[i] = b_lo[i]; bf[8 + i] = b_hi[i];
        }
        acc = __builtin_amdgcn_wmma_f32_16x16x32_f16(false, af, false, bf,
                                                     (short)0, acc, false, false);
    }
    // D layout: VGPR r -> row m0 + r + 8*hi, col n0 + lr
    float* crow = C + (size_t)(m0 + hi * 8) * HH + n0 + lr;
    if (m0 + 16 <= nrows) {            // uniform fast path: full tile, no branches
#pragma unroll
        for (int r = 0; r < 8; ++r) crow[(size_t)r * HH] = acc[r];
    } else {                           // last partial tile only
#pragma unroll
        for (int r = 0; r < 8; ++r) {
            int mm = m0 + r + hi * 8;
            if (mm < nrows) C[(size_t)mm * HH + n0 + lr] = acc[r];
        }
    }
}

// ---------------- fused bias + LayerNorm + ReLU (+residual), dual f32/f16 output ----------------
__global__ __launch_bounds__(128) void k_ln_relu(const float* __restrict__ in,
                                                 const float* __restrict__ bias,
                                                 const float* __restrict__ gamma,
                                                 const float* __restrict__ beta,
                                                 const float* __restrict__ resid,
                                                 float* __restrict__ h32,
                                                 _Float16* __restrict__ h16) {
    const int row = blockIdx.x, c = threadIdx.x;
    float v = in[(size_t)row * HH + c] + bias[c];
    float s = v, s2 = v * v;
#pragma unroll
    for (int off = 16; off > 0; off >>= 1) {
        s  += __shfl_xor(s,  off, 32);
        s2 += __shfl_xor(s2, off, 32);
    }
    __shared__ float ps[4], ps2[4];
    int w = c >> 5;
    if ((c & 31) == 0) { ps[w] = s; ps2[w] = s2; }
    __syncthreads();
    float ts  = ps[0]  + ps[1]  + ps[2]  + ps[3];
    float ts2 = ps2[0] + ps2[1] + ps2[2] + ps2[3];
    float mu  = ts * (1.0f / HH);
    float var = ts2 * (1.0f / HH) - mu * mu;
    float o = (v - mu) * rsqrtf(var + 1e-5f) * gamma[c] + beta[c];
    o = fmaxf(o, 0.0f);
    if (resid) o += resid[(size_t)row * HH + c];
    h32[(size_t)row * HH + c] = o;
    h16[(size_t)row * HH + c] = (_Float16)o;
}

// ---------------- edge scatter: one wave per edge, float4 per lane ----------------
// grid-stride over edges with software prefetch of the next edge's source row
__global__ __launch_bounds__(256) void k_scatter(const int* __restrict__ src,
                                                 const int* __restrict__ dst,
                                                 const float* __restrict__ dinv,
                                                 const float* __restrict__ hw,
                                                 float* __restrict__ agg, int E) {
    const int lane = threadIdx.x & 31;
    const int wid  = blockIdx.x * 8 + (threadIdx.x >> 5);
    const int nw   = gridDim.x * 8;
    for (int e = wid; e < E; e += nw) {
        int en = e + nw;
        if (en < E) {   // hide random-gather latency of the next iteration
            int sn = src[en];
            __builtin_prefetch(hw + (size_t)sn * HH + lane * 4, 0, 1);
        }
        int s = src[e], d = dst[e];
        float w = dinv[s] * dinv[d];
        const float4 v = *(const float4*)(hw + (size_t)s * HH + lane * 4);
        float* basep = agg + (size_t)d * HH + lane * 4;
        __hip_atomic_fetch_add(basep + 0, v.x * w, __ATOMIC_RELAXED, __HIP_MEMORY_SCOPE_AGENT);
        __hip_atomic_fetch_add(basep + 1, v.y * w, __ATOMIC_RELAXED, __HIP_MEMORY_SCOPE_AGENT);
        __hip_atomic_fetch_add(basep + 2, v.z * w, __ATOMIC_RELAXED, __HIP_MEMORY_SCOPE_AGENT);
        __hip_atomic_fetch_add(basep + 3, v.w * w, __ATOMIC_RELAXED, __HIP_MEMORY_SCOPE_AGENT);
    }
}

// ---------------- graph pooling ----------------
// batch is sorted: run-length accumulate POOL_ROWS rows in registers, flush per group
// change (~16x fewer atomics). h >= 0 post-ReLU/residual, so int-max / 0-init works.
__global__ __launch_bounds__(128) void k_pool(const float* __restrict__ h, const int* __restrict__ batch,
                                              float* __restrict__ sum, int* __restrict__ mx,
                                              int* __restrict__ cnt, int n) {
    const int r0 = blockIdx.x * POOL_ROWS;
    const int c  = threadIdx.x;
    int   gcur = batch[r0];
    float sacc = 0.0f, macc = 0.0f;
    int   ccnt = 0;
    for (int i = 0; i < POOL_ROWS; ++i) {
        int row = r0 + i;
        if (row >= n) break;
        int g = batch[row];
        if (g != gcur) {
            __hip_atomic_fetch_add(&sum[gcur * HH + c], sacc, __ATOMIC_RELAXED, __HIP_MEMORY_SCOPE_AGENT);
            atomicMax(&mx[gcur * HH + c], __float_as_int(macc));
            if (c == 0) atomicAdd(&cnt[gcur], ccnt);
            sacc = 0.0f; macc = 0.0f; ccnt = 0; gcur = g;
        }
        float v = h[(size_t)row * HH + c];
        sacc += v; macc = fmaxf(macc, v); ++ccnt;
    }
    __hip_atomic_fetch_add(&sum[gcur * HH + c], sacc, __ATOMIC_RELAXED, __HIP_MEMORY_SCOPE_AGENT);
    atomicMax(&mx[gcur * HH + c], __float_as_int(macc));
    if (c == 0) atomicAdd(&cnt[gcur], ccnt);
}

// ---------------- head (G=64 rows; tiny) ----------------
__global__ __launch_bounds__(128) void k_head1(const float* __restrict__ gf,
                                               const float* __restrict__ W1, const float* __restrict__ b1,
                                               const float* __restrict__ W2, const float* __restrict__ b2,
                                               const float* __restrict__ sum, const int* __restrict__ mx,
                                               const int* __restrict__ cnt, float* __restrict__ z) {
    const int g = blockIdx.x, c = threadIdx.x;
    __shared__ float t1[64];
    if (c < 64) {
        float a = b1[c];
#pragma unroll
        for (int j = 0; j < 4; ++j) a += gf[g * 4 + j] * W1[j * 64 + c];
        t1[c] = fmaxf(a, 0.0f);
    }
    __syncthreads();
    float ge = b2[c];
    for (int j = 0; j < 64; ++j) ge += t1[j] * W2[j * HH + c];
    float cn = fmaxf((float)cnt[g], 1.0f);
    float s  = sum[g * HH + c];
    z[g * 512 + c]       = s / cn;
    z[g * 512 + 128 + c] = __int_as_float(mx[g * HH + c]);
    z[g * 512 + 256 + c] = s;
    z[g * 512 + 384 + c] = ge;
}

__global__ __launch_bounds__(256) void k_head2(const float* __restrict__ z,
                                               const float* __restrict__ W1, const float* __restrict__ b1,
                                               const float* __restrict__ gamma, const float* __restrict__ beta,
                                               float* __restrict__ z1) {
    const int g = blockIdx.x, c = threadIdx.x;
    __shared__ float zr[512];
    for (int j = c; j < 512; j += 256) zr[j] = z[g * 512 + j];
    __syncthreads();
    float a = b1[c];
    for (int j = 0; j < 512; ++j) a += zr[j] * W1[j * 256 + c];
    float s = a, s2 = a * a;
#pragma unroll
    for (int off = 16; off > 0; off >>= 1) {
        s  += __shfl_xor(s,  off, 32);
        s2 += __shfl_xor(s2, off, 32);
    }
    __shared__ float ps[8], ps2[8];
    int w = c >> 5;
    if ((c & 31) == 0) { ps[w] = s; ps2[w] = s2; }
    __syncthreads();
    float ts = 0.f, ts2 = 0.f;
#pragma unroll
    for (int i = 0; i < 8; ++i) { ts += ps[i]; ts2 += ps2[i]; }
    float mu = ts / 256.f, var = ts2 / 256.f - mu * mu;
    float o = (a - mu) * rsqrtf(var + 1e-5f) * gamma[c] + beta[c];
    z1[g * 256 + c] = fmaxf(o, 0.0f);
}

__global__ __launch_bounds__(128) void k_head3(const float* __restrict__ z1,
                                               const float* __restrict__ W2, const float* __restrict__ b2,
                                               float* __restrict__ z2) {
    const int g = blockIdx.x, c = threadIdx.x;
    float a = b2[c];
    for (int j = 0; j < 256; ++j) a += z1[g * 256 + j] * W2[j * 128 + c];
    z2[g * 128 + c] = fmaxf(a, 0.0f);
}

__global__ __launch_bounds__(32) void k_head4(const float* __restrict__ z2,
                                              const float* __restrict__ W3, const float* __restrict__ b3,
                                              float* __restrict__ out) {
    const int g = blockIdx.x, c = threadIdx.x;
    float a = b3[c];
    for (int j = 0; j < 128; ++j) a += z2[g * 128 + j] * W3[j * 32 + c];
    float m = a;
#pragma unroll
    for (int off = 16; off > 0; off >>= 1) m = fmaxf(m, __shfl_xor(m, off, 32));
    float e = __expf(a - m);
    float s = e;
#pragma unroll
    for (int off = 16; off > 0; off >>= 1) s += __shfl_xor(s, off, 32);
    out[g * 32 + c] = e / s;
}

// ---------------- host orchestration ----------------
extern "C" void kernel_launch(void* const* d_in, const int* in_sizes, int n_in,
                              void* d_out, int out_size, void* d_ws, size_t ws_size,
                              hipStream_t stream) {
    const float* x       = (const float*)d_in[0];
    const int*   ei      = (const int*)d_in[1];
    const int*   batch   = (const int*)d_in[2];
    const float* gf      = (const float*)d_in[3];
    const float* enc_W   = (const float*)d_in[4];
    const float* enc_b   = (const float*)d_in[5];
    const float* enc_g   = (const float*)d_in[6];
    const float* enc_be  = (const float*)d_in[7];
    const float* conv_W  = (const float*)d_in[8];
    const float* conv_b  = (const float*)d_in[9];
    const float* norm_g  = (const float*)d_in[10];
    const float* norm_be = (const float*)d_in[11];
    const float* gp_W1   = (const float*)d_in[12];
    const float* gp_b1   = (const float*)d_in[13];
    const float* gp_W2   = (const float*)d_in[14];
    const float* gp_b2   = (const float*)d_in[15];
    const float* dec_W1  = (const float*)d_in[16];
    const float* dec_b1  = (const float*)d_in[17];
    const float* dec_g   = (const float*)d_in[18];
    const float* dec_be  = (const float*)d_in[19];
    const float* dec_W2  = (const float*)d_in[20];
    const float* dec_b2  = (const float*)d_in[21];
    const float* dec_W3  = (const float*)d_in[22];
    const float* dec_b3  = (const float*)d_in[23];
    float* out = (float*)d_out;

    const int N = in_sizes[2];          // batch has N entries
    const int E = in_sizes[1] / 2;
    const int* src = ei;
    const int* dst = ei + E;

    // workspace carve-up (256B aligned)
    char* base = (char*)d_ws;
    size_t off = 0;
    auto carve = [&](size_t bytes) { void* p = base + off; off += (bytes + 255) & ~(size_t)255; return p; };
    float*    h32    = (float*)   carve((size_t)N * HH * 4);
    _Float16* h16    = (_Float16*)carve((size_t)N * HH * 2);
    float*    hw     = (float*)   carve((size_t)N * HH * 4);
    float*    agg    = (float*)   carve((size_t)N * HH * 4);
    _Float16* x16    = (_Float16*)carve((size_t)N * 64 * 2);
    int*      deg    = (int*)     carve((size_t)N * 4);
    float*    dinv   = (float*)   carve((size_t)N * 4);
    _Float16* encWt  = (_Float16*)carve(128 * 64 * 2);
    _Float16* convWt = (_Float16*)carve(4 * 128 * 128 * 2);
    float*    psum   = (float*)   carve(GG * HH * 4);
    int*      pmax   = (int*)     carve(GG * HH * 4);
    int*      cnt    = (int*)     carve(GG * 4);
    float*    z      = (float*)   carve(GG * 512 * 4);
    float*    z1     = (float*)   carve(GG * 256 * 4);
    float*    z2     = (float*)   carve(GG * 128 * 4);
    (void)ws_size; (void)n_in; (void)out_size;

    const int T = 256;
    auto blocks = [](long n, int t) { return (int)((n + t - 1) / t); };

    // prep: f16 conversions + weight transposes
    k_f32_to_f16<<<blocks((long)N * 64, T), T, 0, stream>>>(x, x16, N * 64);
    k_transpose_f16<<<blocks(64 * 128, T), T, 0, stream>>>(enc_W, encWt, 64, 128);
    for (int l = 0; l < 4; ++l)
        k_transpose_f16<<<blocks(128 * 128, T), T, 0, stream>>>(conv_W + l * 128 * 128,
                                                                convWt + l * 128 * 128, 128, 128);
    // degree / symmetric norm
    k_zero_i32<<<blocks(N, T), T, 0, stream>>>(deg, N);
    k_deg_accum<<<blocks(E, T), T, 0, stream>>>(dst, deg, E);
    k_make_dinv<<<blocks(N, T), T, 0, stream>>>(deg, dinv, N);

    const int mblk = (N + 15) / 16;
    // encoder: x @ enc_W -> LN -> ReLU
    k_gemm_wmma<64><<<mblk, 256, 0, stream>>>(x16, encWt, hw, N);
    k_ln_relu<<<N, 128, 0, stream>>>(hw, enc_b, enc_g, enc_be, nullptr, h32, h16);

    // 4 GCN layers
    const int sblocks = 8192;           // grid-stride scatter: ~12 edges per wave
    for (int l = 0; l < 4; ++l) {
        k_gemm_wmma<128><<<mblk, 256, 0, stream>>>(h16, convWt + l * 128 * 128, hw, N);
        k_self_loop<<<blocks((long)N * HH, T), T, 0, stream>>>(hw, dinv, agg, N);
        k_scatter<<<sblocks, 256, 0, stream>>>(src, dst, dinv, hw, agg, E);
        k_ln_relu<<<N, 128, 0, stream>>>(agg, conv_b + l * HH, norm_g + l * HH,
                                         norm_be + l * HH, h32, h32, h16);
    }

    // pooling
    k_zero_i32<<<blocks(GG * HH, T), T, 0, stream>>>((int*)psum, GG * HH);
    k_zero_i32<<<blocks(GG * HH, T), T, 0, stream>>>(pmax, GG * HH);
    k_zero_i32<<<1, GG, 0, stream>>>(cnt, GG);
    k_pool<<<blocks(N, POOL_ROWS), 128, 0, stream>>>(h32, batch, psum, pmax, cnt, N);

    // head
    k_head1<<<GG, 128, 0, stream>>>(gf, gp_W1, gp_b1, gp_W2, gp_b2, psum, pmax, cnt, z);
    k_head2<<<GG, 256, 0, stream>>>(z, dec_W1, dec_b1, dec_g, dec_be, z1);
    k_head3<<<GG, 128, 0, stream>>>(z1, dec_W2, dec_b2, z2);
    k_head4<<<GG, 32, 0, stream>>>(z2, dec_W3, dec_b3, out);
}
